// ProtoGraphLayer_62036507623829
// MI455X (gfx1250) — matrix operand order repaired and checked by reading, not compiled, not run on hardware
//
#include <hip/hip_runtime.h>

// ---------------------------------------------------------------------------
// Problem constants (from reference)
// ---------------------------------------------------------------------------
static constexpr int   N_  = 8000;
static constexpr int   D_  = 256;
static constexpr int   H_  = 4;
static constexpr int   E_  = 128000;
static constexpr size_t NF  = (size_t)N_ * D_;        // 2,048,000
static constexpr size_t NHD = (size_t)N_ * H_ * D_;   // 8,192,000
static constexpr int   EN  = E_ + N_;                 // edges incl self loops
static constexpr unsigned CAP = 1u << 19;             // candidate buffer (>= E + slack)

// Workspace layout (bytes, all 16B aligned)
static constexpr size_t o_h0   = 0;                       // N*D f32
static constexpr size_t o_h1   = o_h0   + NF * 4;         // N*D f32
static constexpr size_t o_xp   = o_h1   + NF * 4;         // N*H*D f32
static constexpr size_t o_acc  = o_xp   + NHD * 4;        // N*H*D f32
static constexpr size_t o_h16  = o_acc  + NHD * 4;        // N*D f16 (A operand)
static constexpr size_t o_wt16 = o_h16  + NF * 2;         // up to 1024*256 f16 (Bt operand)
static constexpr size_t o_asrc = o_wt16 + (size_t)D_ * H_ * D_ * 2;
static constexpr size_t o_adst = o_asrc + (size_t)N_ * H_ * 4;
static constexpr size_t o_m    = o_adst + (size_t)N_ * H_ * 4;
static constexpr size_t o_z    = o_m    + (size_t)N_ * H_ * 4;
static constexpr size_t o_e    = o_z    + (size_t)N_ * H_ * 4;   // (E+N)*H f32
static constexpr size_t o_deg  = o_e    + (size_t)EN * H_ * 4;
static constexpr size_t o_dinv = o_deg  + (size_t)N_ * 4;
static constexpr size_t o_ench = o_dinv + (size_t)N_ * 4;        // N*D f32
static constexpr size_t o_enct = o_ench + NF * 4;                // N*D f32
static constexpr size_t o_eh16 = o_enct + NF * 4;                // N*D f16
static constexpr size_t o_t016 = o_eh16 + NF * 2;
static constexpr size_t o_t116 = o_t016 + NF * 2;
static constexpr size_t o_wl16 = o_t116 + NF * 2;                // D*D f16
static constexpr size_t o_hist1= o_wl16 + (size_t)D_ * D_ * 2;
static constexpr size_t o_hist2= o_hist1 + 4096 * 4;
static constexpr size_t o_sel  = o_hist2 + 4096 * 4;             // 64B
static constexpr size_t o_ctr  = o_sel  + 64;                    // 64B
static constexpr size_t o_cand = o_ctr  + 64;                    // CAP u64
static constexpr size_t WS_NEED= o_cand + (size_t)CAP * 8;

// ---------------------------------------------------------------------------
// Vector types for WMMA
// ---------------------------------------------------------------------------
typedef __attribute__((ext_vector_type(16))) _Float16 v16h;
typedef __attribute__((ext_vector_type(8)))  _Float16 v8h;
typedef __attribute__((ext_vector_type(8)))  float    v8f;
typedef __attribute__((ext_vector_type(4)))  int      v4i;

// Per-lane A/B fragment for v_wmma_f32_16x16x32_f16.
// base points at (row, k0 + half*8); elems 0..7 = K kb..kb+7, 8..15 = K kb+16..kb+23.
__device__ inline v16h load_frag(const _Float16* base) {
  const v8h* p = (const v8h*)base;
  v8h lo = p[0];
  v8h hi = p[2];
  return __builtin_shufflevector(lo, hi, 0,1,2,3,4,5,6,7,8,9,10,11,12,13,14,15);
}

__device__ inline v8f wmma32(v16h a, v16h b, v8f c) {
  return __builtin_amdgcn_wmma_f32_16x16x32_f16(false, a, false, b, (short)0, c,
                                                false, false);
}

// ---------------------------------------------------------------------------
// CDNA5 async global->LDS copy (ASYNCcnt path), with safe fallback.
// Probe round 2 revealed the builtin signature: arg0 = int4 AS(1)*, arity 4.
// ---------------------------------------------------------------------------
#if defined(__has_builtin)
#  if __has_builtin(__builtin_amdgcn_global_load_async_to_lds_b128)
#    define ASYNC_LDS 1
#  endif
#endif
#ifndef ASYNC_LDS
#  define ASYNC_LDS 0
#endif

#if ASYNC_LDS
typedef __attribute__((address_space(1))) v4i as1_v4i;   // global (__device__)
typedef __attribute__((address_space(3))) v4i as3_v4i;   // LDS (__shared__)
__device__ inline void async_cp16(const _Float16* g, _Float16* l) {
  __builtin_amdgcn_global_load_async_to_lds_b128((as1_v4i*)g, (as3_v4i*)l, 0, 0);
}
__device__ inline void wait_async0() {
#if __has_builtin(__builtin_amdgcn_s_wait_asynccnt)
  __builtin_amdgcn_s_wait_asynccnt(0);
#else
  asm volatile("s_wait_asynccnt 0x0" ::: "memory");
#endif
}
#endif

// ---------------------------------------------------------------------------
// Elementwise / utility kernels
// ---------------------------------------------------------------------------
__global__ void fill_f32(float* p, float v, int n) {
  int i = blockIdx.x * 256 + threadIdx.x;
  if (i < n) p[i] = v;
}
__global__ void fill_u32(unsigned* p, unsigned v, int n) {
  int i = blockIdx.x * 256 + threadIdx.x;
  if (i < n) p[i] = v;
}
__global__ void fill_u64(unsigned long long* p, unsigned long long v, int n) {
  int i = blockIdx.x * 256 + threadIdx.x;
  if (i < n) p[i] = v;
}
__global__ void cast_f16(const float* x, _Float16* y, int n) {
  int i = blockIdx.x * 256 + threadIdx.x;
  if (i < n) y[i] = (_Float16)x[i];
}
// W[rows, cols] row-major -> Wt[cols, rows] f16 (Bt operand for the GEMM)
__global__ void transpose_cast(const float* W, _Float16* Wt, int rows, int cols) {
  int i = blockIdx.x * 256 + threadIdx.x;
  if (i >= rows * cols) return;
  int r = i / cols, c = i % cols;
  Wt[(size_t)c * rows + r] = (_Float16)W[i];
}
__global__ void add_bias(float* x, const float* b, int n, int d) {
  int i = blockIdx.x * 256 + threadIdx.x;
  if (i < n * d) x[i] += b[i % d];
}

// ---------------------------------------------------------------------------
// LayerNorm (eps=1e-12, biased variance) — one wave per row
// ---------------------------------------------------------------------------
__global__ void __launch_bounds__(256) layernorm_k(const float* x, const float* g,
                                                   const float* b, float* y) {
  int wave = (blockIdx.x * 256 + threadIdx.x) >> 5;
  int lane = threadIdx.x & 31;
  if (wave >= N_) return;
  const float* row = x + (size_t)wave * D_;
  float s = 0.f, s2 = 0.f;
  for (int c = lane; c < D_; c += 32) { float v = row[c]; s += v; s2 += v * v; }
  for (int off = 16; off; off >>= 1) {
    s  += __shfl_xor(s, off, 32);
    s2 += __shfl_xor(s2, off, 32);
  }
  float mu  = s / D_;
  float var = s2 / D_ - mu * mu;
  float inv = rsqrtf(var + 1e-12f);
  float* yr = y + (size_t)wave * D_;
  for (int c = lane; c < D_; c += 32) yr[c] = (row[c] - mu) * inv * g[c] + b[c];
}

// ---------------------------------------------------------------------------
// Dense feature GEMM (small K-panels): C[M,Nc] = A[M,K] * Bt[Nc,K]^T
// One wave per 16x16 tile; pipelines loads against WMMA well.
// ---------------------------------------------------------------------------
__global__ void __launch_bounds__(128) gemm_f16_wmma(const _Float16* __restrict__ A,
                                                     const _Float16* __restrict__ Bt,
                                                     float* __restrict__ C,
                                                     int M, int K, int Nc) {
  int gwave = (blockIdx.x * 128 + threadIdx.x) >> 5;
  int lane  = threadIdx.x & 31;
  int tilesN = Nc >> 4;
  int tiles  = (M >> 4) * tilesN;
  if (gwave >= tiles) return;
  int tm = (gwave / tilesN) << 4;
  int tn = (gwave % tilesN) << 4;
  int half_ = lane >> 4, lrow = lane & 15;
  const _Float16* arow = A  + (size_t)(tm + lrow) * K + half_ * 8;
  const _Float16* brow = Bt + (size_t)(tn + lrow) * K + half_ * 8;
  v8f acc = {};
  for (int k0 = 0; k0 < K; k0 += 32) {
    __builtin_prefetch(arow + k0 + 64, 0, 0);
    __builtin_prefetch(brow + k0 + 64, 0, 0);
    v16h a = load_frag(arow + k0);
    v16h b = load_frag(brow + k0);
    acc = wmma32(a, b, acc);
  }
  float* crow = C + (size_t)(tm + half_ * 8) * Nc + tn + lrow;
#pragma unroll
  for (int r = 0; r < 8; ++r) crow[(size_t)r * Nc] = acc[r];
}

// ---------------------------------------------------------------------------
// GAT pieces
// ---------------------------------------------------------------------------
__global__ void __launch_bounds__(128) attn_coef(const float* xp, const float* a,
                                                 float* out, int heads) {
  int wave = (blockIdx.x * 128 + threadIdx.x) >> 5;
  int lane = threadIdx.x & 31;
  if (wave >= N_ * heads) return;
  int n = wave / heads, h = wave % heads;
  const float* xr = xp + ((size_t)n * heads + h) * D_;
  const float* ar = a + (size_t)h * D_;
  float s = 0.f;
  for (int c = lane; c < D_; c += 32) s += xr[c] * ar[c];
  for (int off = 16; off; off >>= 1) s += __shfl_xor(s, off, 32);
  if (lane == 0) out[wave] = s;
}

__device__ inline void atomicMaxFloat(float* addr, float val) {
  unsigned* ua = (unsigned*)addr;
  unsigned old = *ua;
  while (__uint_as_float(old) < val) {
    unsigned assumed = old;
    old = atomicCAS(ua, assumed, __float_as_uint(val));
    if (old == assumed) break;
  }
}

__device__ inline void edge_sd(const long long* src, const long long* dst,
                               int e, int& s, int& d) {
  if (e < E_) { s = (int)src[e]; d = (int)dst[e]; }
  else        { s = e - E_;      d = e - E_; }
}

__global__ void edge_alpha_max(const long long* src, const long long* dst,
                               const float* asrc, const float* adst,
                               float* m, int heads) {
  int i = blockIdx.x * 256 + threadIdx.x;
  if (i >= EN * heads) return;
  int e = i / heads, h = i % heads, s, d;
  edge_sd(src, dst, e, s, d);
  float a = asrc[s * heads + h] + adst[d * heads + h];
  a = a > 0.f ? a : 0.2f * a;                  // leaky_relu(0.2)
  atomicMaxFloat(&m[d * heads + h], a);
}

__global__ void edge_exp_sum(const long long* src, const long long* dst,
                             const float* asrc, const float* adst,
                             const float* m, float* ebuf, float* z, int heads) {
  int i = blockIdx.x * 256 + threadIdx.x;
  if (i >= EN * heads) return;
  int e = i / heads, h = i % heads, s, d;
  edge_sd(src, dst, e, s, d);
  float a = asrc[s * heads + h] + adst[d * heads + h];
  a = a > 0.f ? a : 0.2f * a;
  float ex = expf(a - m[d * heads + h]);
  ebuf[i] = ex;
  atomicAdd(&z[d * heads + h], ex);
}

__global__ void __launch_bounds__(256) edge_msg_acc(const long long* src,
                                                    const long long* dst,
                                                    const float* xp,
                                                    const float* ebuf, const float* z,
                                                    float* acc, int heads) {
  int e = blockIdx.x, s, d;
  edge_sd(src, dst, e, s, d);
  int HD = heads * D_;
  for (int c = threadIdx.x; c < HD; c += 256) {
    int h = c / D_;
    float w = ebuf[(size_t)e * heads + h] / z[d * heads + h];
    atomicAdd(&acc[(size_t)d * HD + c], xp[(size_t)s * HD + c] * w);
  }
}

// faithful reshape(N, D, H).mean(-1): out[n,c] = mean_j acc[n, c*H + j]
__global__ void head_mean(const float* acc, float* out) {
  int i = blockIdx.x * 256 + threadIdx.x;
  if (i >= (int)NF) return;
  int n = i / D_, c = i % D_;
  const float* a = acc + (size_t)n * (H_ * D_) + c * H_;
  out[i] = 0.25f * (a[0] + a[1] + a[2] + a[3]);
}

// ---------------------------------------------------------------------------
// GCN pieces
// ---------------------------------------------------------------------------
__global__ void gcn_deg(const long long* dst, float* deg) {
  int e = blockIdx.x * 256 + threadIdx.x;
  if (e < E_) atomicAdd(&deg[(int)dst[e]], 1.0f);
}
__global__ void gcn_dinv(const float* deg, float* dinv) {
  int i = blockIdx.x * 256 + threadIdx.x;
  if (i < N_) dinv[i] = deg[i] > 0.f ? rsqrtf(deg[i]) : 0.f;
}
__global__ void __launch_bounds__(256) gcn_acc(const long long* src, const long long* dst,
                                               const float* xp, const float* dinv,
                                               float* out) {
  int e = blockIdx.x;
  int s = (int)src[e], d = (int)dst[e];
  float w = dinv[s] * dinv[d];
  for (int c = threadIdx.x; c < D_; c += 256)
    atomicAdd(&out[(size_t)d * D_ + c], xp[(size_t)s * D_ + c] * w);
}

// ---------------------------------------------------------------------------
// Score GEMM: LDS-block-tiled, 64x64 output per 256-thread block (8 waves).
// Each wave owns two 16x16 subtiles (rows +0/+32) and both score matrices:
// 8 WMMAs per 64-deep K-chunk with 4x LDS reuse of every fragment.
// ---------------------------------------------------------------------------
static constexpr int TILES_N   = N_ / 16;          // 500
static constexpr int BLOCKS_M  = TILES_N / 4;      // 125 (64-wide block tiles)
static constexpr int SCORE_BLOCKS = BLOCKS_M * BLOCKS_M;
static constexpr int KC  = 64;                     // K chunk depth
static constexpr int STR = 72;                     // padded LDS row stride (halves)

struct BlockAcc {
  v8f s0a, s1a, s0b, s1b;
  int tmA, tmB, tn;
};

__device__ inline BlockAcc score_block(const _Float16* __restrict__ EH,
                                       const _Float16* __restrict__ T0,
                                       const _Float16* __restrict__ T1,
                                       _Float16* shA, _Float16* shB0, _Float16* shB1) {
  int bid = blockIdx.x;
  int tm0 = (bid / BLOCKS_M) * 64;
  int tn0 = (bid % BLOCKS_M) * 64;
  int t = threadIdx.x;
  int wave = t >> 5, lane = t & 31;
  int half_ = lane >> 4, lrow = lane & 15;
  int tmLa = (wave >> 2) * 16;       // 0 or 16
  int tmLb = tmLa + 32;              // 32 or 48
  int tnL  = (wave & 3) * 16;        // 0..48
  int lr = t >> 2;                   // 0..63 : row loaded by this thread
  int lc = (t & 3) * 16;             // 0,16,32,48 : 16-half column slab

  BlockAcc R;
  R.s0a = (v8f){}; R.s1a = (v8f){}; R.s0b = (v8f){}; R.s1b = (v8f){};

  for (int kc = 0; kc < D_; kc += KC) {
    const _Float16* gA = EH + (size_t)(tm0 + lr) * D_ + kc + lc;
    const _Float16* g0 = T0 + (size_t)(tn0 + lr) * D_ + kc + lc;
    const _Float16* g1 = T1 + (size_t)(tn0 + lr) * D_ + kc + lc;
    _Float16* dA = shA  + lr * STR + lc;
    _Float16* d0 = shB0 + lr * STR + lc;
    _Float16* d1 = shB1 + lr * STR + lc;
#if ASYNC_LDS
    async_cp16(gA, dA);
    async_cp16(g0, d0);
    async_cp16(g1, d1);
    async_cp16(gA + 8, dA + 8);
    async_cp16(g0 + 8, d0 + 8);
    async_cp16(g1 + 8, d1 + 8);
    wait_async0();
#else
    *(v8h*)dA = *(const v8h*)gA;
    *(v8h*)d0 = *(const v8h*)g0;
    *(v8h*)d1 = *(const v8h*)g1;
    *(v8h*)(dA + 8) = *(const v8h*)(gA + 8);
    *(v8h*)(d0 + 8) = *(const v8h*)(g0 + 8);
    *(v8h*)(d1 + 8) = *(const v8h*)(g1 + 8);
#endif
    __syncthreads();
#pragma unroll
    for (int k0 = 0; k0 < KC; k0 += 32) {
      int kb = k0 + half_ * 8;
      v16h aA = load_frag(shA  + (tmLa + lrow) * STR + kb);
      v16h aB = load_frag(shA  + (tmLb + lrow) * STR + kb);
      v16h b0 = load_frag(shB0 + (tnL + lrow) * STR + kb);
      v16h b1 = load_frag(shB1 + (tnL + lrow) * STR + kb);
      R.s0a = wmma32(aA, b0, R.s0a);
      R.s1a = wmma32(aA, b1, R.s1a);
      R.s0b = wmma32(aB, b0, R.s0b);
      R.s1b = wmma32(aB, b1, R.s1b);
    }
    __syncthreads();
  }
  R.tmA = tm0 + tmLa + half_ * 8;    // element r of s*a -> row tmA + r
  R.tmB = tm0 + tmLb + half_ * 8;
  R.tn  = tn0 + tnL + lrow;
  return R;
}

__device__ inline unsigned key_of(float s0, float s1) {
  float v = (s1 > s0) ? s1 : -__builtin_inff();
  unsigned u = __float_as_uint(v);
  return (u >> 31) ? ~u : (u | 0x80000000u);
}

__global__ void __launch_bounds__(256) score_hist1(const _Float16* EH, const _Float16* T0,
                                                   const _Float16* T1, unsigned* hist) {
  __shared__ _Float16 shA[64 * STR], shB0[64 * STR], shB1[64 * STR];
  __shared__ unsigned sh[4096];
  for (int i = threadIdx.x; i < 4096; i += 256) sh[i] = 0;
  __syncthreads();
  BlockAcc R = score_block(EH, T0, T1, shA, shB0, shB1);
#pragma unroll
  for (int r = 0; r < 8; ++r) {
    atomicAdd(&sh[key_of(R.s0a[r], R.s1a[r]) >> 20], 1u);
    atomicAdd(&sh[key_of(R.s0b[r], R.s1b[r]) >> 20], 1u);
  }
  __syncthreads();
  for (int i = threadIdx.x; i < 4096; i += 256)
    if (sh[i]) atomicAdd(&hist[i], sh[i]);
}

__global__ void select1(const unsigned* hist, unsigned* sel) {
  unsigned cum = 0;
  for (int b = 4095; b >= 0; --b) {
    unsigned h = hist[b];
    if (cum + h >= (unsigned)E_) { sel[0] = (unsigned)b; sel[1] = cum; return; }
    cum += h;
  }
  sel[0] = 0; sel[1] = 0;
}

__global__ void __launch_bounds__(256) score_hist2(const _Float16* EH, const _Float16* T0,
                                                   const _Float16* T1, const unsigned* sel,
                                                   unsigned* hist) {
  __shared__ _Float16 shA[64 * STR], shB0[64 * STR], shB1[64 * STR];
  __shared__ unsigned sh[4096];
  for (int i = threadIdx.x; i < 4096; i += 256) sh[i] = 0;
  __syncthreads();
  unsigned b1 = sel[0];
  BlockAcc R = score_block(EH, T0, T1, shA, shB0, shB1);
#pragma unroll
  for (int r = 0; r < 8; ++r) {
    unsigned ka = key_of(R.s0a[r], R.s1a[r]);
    unsigned kb = key_of(R.s0b[r], R.s1b[r]);
    if ((ka >> 20) == b1) atomicAdd(&sh[(ka >> 8) & 0xFFFu], 1u);
    if ((kb >> 20) == b1) atomicAdd(&sh[(kb >> 8) & 0xFFFu], 1u);
  }
  __syncthreads();
  for (int i = threadIdx.x; i < 4096; i += 256)
    if (sh[i]) atomicAdd(&hist[i], sh[i]);
}

__global__ void select2(const unsigned* hist2, unsigned* sel) {
  unsigned rem = (unsigned)E_ - sel[1];
  unsigned cum = 0;
  for (int b = 4095; b >= 0; --b) {
    unsigned h = hist2[b];
    if (cum + h >= rem) { sel[2] = (sel[0] << 12) | (unsigned)b; return; }
    cum += h;
  }
  sel[2] = sel[0] << 12;
}

__global__ void __launch_bounds__(256) score_collect(const _Float16* EH, const _Float16* T0,
                                                     const _Float16* T1, const unsigned* sel,
                                                     unsigned long long* cand, unsigned* ctr) {
  __shared__ _Float16 shA[64 * STR], shB0[64 * STR], shB1[64 * STR];
  unsigned thr = sel[2];
  BlockAcc R = score_block(EH, T0, T1, shA, shB0, shB1);
#pragma unroll
  for (int r = 0; r < 8; ++r) {
    unsigned ka = key_of(R.s0a[r], R.s1a[r]);
    if ((ka >> 8) >= thr) {
      unsigned idx = (unsigned)(R.tmA + r) * (unsigned)N_ + (unsigned)R.tn;
      unsigned pos = atomicAdd(ctr, 1u);
      if (pos < CAP)
        cand[pos] = ((unsigned long long)ka << 32) | (unsigned long long)(~idx);
    }
    unsigned kb = key_of(R.s0b[r], R.s1b[r]);
    if ((kb >> 8) >= thr) {
      unsigned idx = (unsigned)(R.tmB + r) * (unsigned)N_ + (unsigned)R.tn;
      unsigned pos = atomicAdd(ctr, 1u);
      if (pos < CAP)
        cand[pos] = ((unsigned long long)kb << 32) | (unsigned long long)(~idx);
    }
  }
}

// Descending bitonic sort on packed (key<<32 | ~idx); zero padding sinks.
__global__ void bitonic_step(unsigned long long* data, unsigned j, unsigned k) {
  unsigned i = blockIdx.x * 256 + threadIdx.x;
  unsigned ixj = i ^ j;
  if (ixj > i && ixj < CAP) {
    unsigned long long a = data[i], b = data[ixj];
    bool desc = ((i & k) == 0);
    if (desc ? (a < b) : (a > b)) { data[i] = b; data[ixj] = a; }
  }
}

__global__ void finalize_k(const unsigned long long* cand, long long* out_edges,
                           float* out_w) {
  int i = blockIdx.x * 256 + threadIdx.x;
  if (i >= E_) return;
  unsigned long long ent = cand[i];
  unsigned key = (unsigned)(ent >> 32);
  unsigned idx = ~(unsigned)(ent & 0xFFFFFFFFull);
  unsigned u = (key & 0x80000000u) ? (key & 0x7FFFFFFFu) : ~key;
  float v = __uint_as_float(u);
  float sg = 1.0f / (1.0f + expf(-v));
  out_edges[i]      = (long long)(idx / (unsigned)N_);
  out_edges[E_ + i] = (long long)(idx % (unsigned)N_);
  out_w[i] = (sg + 1.0f) * 0.5f;
}

// ---------------------------------------------------------------------------
// Host-side helpers
// ---------------------------------------------------------------------------
static inline int cdiv(int a, int b) { return (a + b - 1) / b; }

static void gemm(const _Float16* A, const _Float16* Bt, float* C,
                 int M, int K, int Nc, hipStream_t st) {
  int tiles = (M / 16) * (Nc / 16);
  gemm_f16_wmma<<<cdiv(tiles, 4), 128, 0, st>>>(A, Bt, C, M, K, Nc);
}

static void run_gat(const float* hin, const float* W, const float* a_s,
                    const float* a_d, const float* bias, int heads,
                    const long long* src, const long long* dst,
                    char* ws, float* out, hipStream_t st) {
  _Float16* h16  = (_Float16*)(ws + o_h16);
  _Float16* wt16 = (_Float16*)(ws + o_wt16);
  float* xp   = (float*)(ws + o_xp);
  float* asrc = (float*)(ws + o_asrc);
  float* adst = (float*)(ws + o_adst);
  float* m    = (float*)(ws + o_m);
  float* z    = (float*)(ws + o_z);
  float* ebuf = (float*)(ws + o_e);
  int HD = heads * D_;

  cast_f16<<<cdiv((int)NF, 256), 256, 0, st>>>(hin, h16, (int)NF);
  transpose_cast<<<cdiv(D_ * HD, 256), 256, 0, st>>>(W, wt16, D_, HD);
  gemm(h16, wt16, xp, N_, D_, HD, st);

  attn_coef<<<cdiv(N_ * heads, 4), 128, 0, st>>>(xp, a_s, asrc, heads);
  attn_coef<<<cdiv(N_ * heads, 4), 128, 0, st>>>(xp, a_d, adst, heads);

  fill_f32<<<cdiv(N_ * heads, 256), 256, 0, st>>>(m, -__builtin_inff(), N_ * heads);
  fill_f32<<<cdiv(N_ * heads, 256), 256, 0, st>>>(z, 0.f, N_ * heads);
  edge_alpha_max<<<cdiv(EN * heads, 256), 256, 0, st>>>(src, dst, asrc, adst, m, heads);
  edge_exp_sum<<<cdiv(EN * heads, 256), 256, 0, st>>>(src, dst, asrc, adst, m, ebuf, z, heads);

  fill_f32<<<cdiv(N_ * HD, 256), 256, 0, st>>>(out, 0.f, N_ * HD);
  edge_msg_acc<<<EN, 256, 0, st>>>(src, dst, xp, ebuf, z, out, heads);
  if (bias) add_bias<<<cdiv(N_ * HD, 256), 256, 0, st>>>(out, bias, N_, HD);
}

// ---------------------------------------------------------------------------
// Entry point
// ---------------------------------------------------------------------------
extern "C" void kernel_launch(void* const* d_in, const int* in_sizes, int n_in,
                              void* d_out, int out_size, void* d_ws, size_t ws_size,
                              hipStream_t stream) {
  (void)in_sizes; (void)n_in; (void)out_size;
  if (ws_size < WS_NEED) return;

  const float*     hidden   = (const float*)d_in[0];
  const long long* eidx     = (const long long*)d_in[1];
  const float*     ln_g     = (const float*)d_in[2];
  const float*     ln_b     = (const float*)d_in[3];
  const float*     W_gat0   = (const float*)d_in[4];
  const float*     att_s0   = (const float*)d_in[5];
  const float*     att_d0   = (const float*)d_in[6];
  const float*     W_gat1   = (const float*)d_in[7];
  const float*     att_s1   = (const float*)d_in[8];
  const float*     att_d1   = (const float*)d_in[9];
  const float*     W_gcn    = (const float*)d_in[10];
  const float*     b_gcn    = (const float*)d_in[11];
  const float*     W_head   = (const float*)d_in[12];
  const float*     att_sh   = (const float*)d_in[13];
  const float*     att_dh   = (const float*)d_in[14];
  const float*     b_head   = (const float*)d_in[15];
  const float*     W_tail   = (const float*)d_in[16];
  const float*     att_st   = (const float*)d_in[17];
  const float*     att_dt   = (const float*)d_in[18];
  const float*     b_tail   = (const float*)d_in[19];
  const float*     W_l0     = (const float*)d_in[20];
  const float*     W_l1     = (const float*)d_in[21];

  const long long* src = eidx;
  const long long* dst = eidx + E_;

  char* ws = (char*)d_ws;
  float* h0   = (float*)(ws + o_h0);
  float* h1   = (float*)(ws + o_h1);
  float* xp   = (float*)(ws + o_xp);
  float* acc  = (float*)(ws + o_acc);
  float* deg  = (float*)(ws + o_deg);
  float* dinv = (float*)(ws + o_dinv);
  float* ench = (float*)(ws + o_ench);
  float* enct = (float*)(ws + o_enct);
  _Float16* h16  = (_Float16*)(ws + o_h16);
  _Float16* wt16 = (_Float16*)(ws + o_wt16);
  _Float16* eh16 = (_Float16*)(ws + o_eh16);
  _Float16* t016 = (_Float16*)(ws + o_t016);
  _Float16* t116 = (_Float16*)(ws + o_t116);
  _Float16* wl16 = (_Float16*)(ws + o_wl16);
  unsigned* hist1 = (unsigned*)(ws + o_hist1);
  unsigned* hist2 = (unsigned*)(ws + o_hist2);
  unsigned* sel   = (unsigned*)(ws + o_sel);
  unsigned* ctr   = (unsigned*)(ws + o_ctr);
  unsigned long long* cand = (unsigned long long*)(ws + o_cand);

  // 1) LayerNorm
  layernorm_k<<<N_ / 8, 256, 0, stream>>>(hidden, ln_g, ln_b, h0);

  // 2) Stacked GATs (H=4) with faithful head-mean reshape
  run_gat(h0, W_gat0, att_s0, att_d0, nullptr, H_, src, dst, ws, acc, stream);
  head_mean<<<cdiv((int)NF, 256), 256, 0, stream>>>(acc, h1);
  run_gat(h1, W_gat1, att_s1, att_d1, nullptr, H_, src, dst, ws, acc, stream);
  head_mean<<<cdiv((int)NF, 256), 256, 0, stream>>>(acc, h0);

  // 3) GCN (no self loops) -> h1 is the returned "h"
  cast_f16<<<cdiv((int)NF, 256), 256, 0, stream>>>(h0, h16, (int)NF);
  transpose_cast<<<cdiv(D_ * D_, 256), 256, 0, stream>>>(W_gcn, wt16, D_, D_);
  gemm(h16, wt16, xp, N_, D_, D_, stream);
  fill_f32<<<cdiv(N_, 256), 256, 0, stream>>>(deg, 0.f, N_);
  gcn_deg<<<cdiv(E_, 256), 256, 0, stream>>>(dst, deg);
  gcn_dinv<<<cdiv(N_, 256), 256, 0, stream>>>(deg, dinv);
  fill_f32<<<cdiv((int)NF, 256), 256, 0, stream>>>(h1, 0.f, (int)NF);
  gcn_acc<<<E_, 256, 0, stream>>>(src, dst, xp, dinv, h1);
  add_bias<<<cdiv((int)NF, 256), 256, 0, stream>>>(h1, b_gcn, N_, D_);
  (void)hipMemcpyAsync(d_out, h1, NF * 4, hipMemcpyDeviceToDevice, stream);

  // 4) head / tail encoders (H=1, with bias)
  run_gat(h1, W_head, att_sh, att_dh, b_head, 1, src, dst, ws, ench, stream);
  run_gat(h1, W_tail, att_st, att_dt, b_tail, 1, src, dst, ws, enct, stream);

  // 5) t0 = enc_tail @ W_l0.T, t1 = enc_tail @ W_l1.T   (Bt = W_l row-major)
  cast_f16<<<cdiv((int)NF, 256), 256, 0, stream>>>(enct, h16, (int)NF);
  cast_f16<<<cdiv(D_ * D_, 256), 256, 0, stream>>>(W_l0, wl16, D_ * D_);
  gemm(h16, wl16, xp, N_, D_, D_, stream);
  cast_f16<<<cdiv((int)NF, 256), 256, 0, stream>>>(xp, t016, (int)NF);
  cast_f16<<<cdiv(D_ * D_, 256), 256, 0, stream>>>(W_l1, wl16, D_ * D_);
  gemm(h16, wl16, xp, N_, D_, D_, stream);
  cast_f16<<<cdiv((int)NF, 256), 256, 0, stream>>>(xp, t116, (int)NF);
  cast_f16<<<cdiv((int)NF, 256), 256, 0, stream>>>(ench, eh16, (int)NF);

  // 6) top-k over masked N x N score matrix (never materialized):
  //    two-level radix histogram -> collect -> bitonic sort -> emit.
  fill_u32<<<16, 256, 0, stream>>>(hist1, 0u, 4096);
  fill_u32<<<16, 256, 0, stream>>>(hist2, 0u, 4096);
  fill_u32<<<1, 256, 0, stream>>>(sel, 0u, 8);
  fill_u32<<<1, 256, 0, stream>>>(ctr, 0u, 1);
  fill_u64<<<cdiv((int)CAP, 256), 256, 0, stream>>>(cand, 0ull, (int)CAP);

  score_hist1<<<SCORE_BLOCKS, 256, 0, stream>>>(eh16, t016, t116, hist1);
  select1<<<1, 1, 0, stream>>>(hist1, sel);
  score_hist2<<<SCORE_BLOCKS, 256, 0, stream>>>(eh16, t016, t116, sel, hist2);
  select2<<<1, 1, 0, stream>>>(hist2, sel);
  score_collect<<<SCORE_BLOCKS, 256, 0, stream>>>(eh16, t016, t116, sel, cand, ctr);

  for (unsigned k = 2; k <= CAP; k <<= 1)
    for (unsigned j = k >> 1; j > 0; j >>= 1)
      bitonic_step<<<CAP / 256, 256, 0, stream>>>(cand, j, k);

  float* outf = (float*)d_out;
  long long* out_edges = (long long*)(outf + NF);
  float* out_w = outf + NF + 4 * (size_t)E_;
  finalize_k<<<cdiv(E_, 256), 256, 0, stream>>>(cand, out_edges, out_w);
}
